// build_unet_36249523978581
// MI455X (gfx1250) — matrix operand order repaired and checked by reading, not compiled, not run on hardware
//
#include <hip/hip_runtime.h>
#include <math.h>

// ---------------------------------------------------------------------------
// CDNA5 (gfx1250) quantum U-Net, round 3.
//  - 3x3 convs: GEMM via v_wmma_f32_16x16x32_f16, K ordered tap*Cin+ci
//    (Cin always pow2 -> shift/mask decode; one tap/bounds-check per 16-run).
//  - Software-pipelined K loop (fragments for tile k+1 fetched before the
//    WMMA on tile k) + two accumulators to break the WMMA RAW chain.
//  - Weight tiles pre-packed to the exact 16-bit A fragment layout; next
//    tile prefetched near (global_prefetch_b8).
//  - Activations pre-converted to f16; decoder concat fused into the
//    ConvTranspose / skip-pack passes.
//  - Quantum bottleneck: 64 workgroups, 512-amp complex statevector in LDS.
// ---------------------------------------------------------------------------

typedef __attribute__((ext_vector_type(16))) _Float16 v16h;
typedef __attribute__((ext_vector_type(8)))  float    v8f;

#define WS_ALIGN(x) (((x) + 255) & ~(size_t)255)

// ---------------------------------------------------------------------------
// Weight packing: W[Cout, Cin, 3, 3] (f32) -> A fragments in GEMM K-order
// k = tap*Cin + ci  (tap = ky*3+kx).  Fragment layout used here (not the
// ISA table k order): lane l: m = l&15, kb = (l>>4)*16;
// element j: k_local = kb + j (contiguous, matches B fragment).
// Stored [Mtiles][Ktiles][lane 0..31][16 halves] contiguous.
// ---------------------------------------------------------------------------
__global__ __launch_bounds__(256) void prep_wfrag(
    const float* __restrict__ w, _Float16* __restrict__ wfrag,
    int Cout, int Cin, int cinShift, int Mtiles, int Ktiles)
{
    int gid = blockIdx.x * blockDim.x + threadIdx.x;
    int total = Mtiles * Ktiles * 32;
    if (gid >= total) return;
    int lane = gid & 31;
    int tile = gid >> 5;
    int kt = tile % Ktiles;
    int mt = tile / Ktiles;
    int co = mt * 16 + (lane & 15);
    int kb = (lane >> 4) * 16;
    int mask = Cin - 1;
    v16h out;
#pragma unroll
    for (int j = 0; j < 16; ++j) {
        int vg = j >> 1, hf = j & 1;
        int kl = kb + j; (void)vg; (void)hf;
        int k  = kt * 32 + kl;
        int tp = k >> cinShift;
        int ci = k & mask;
        float v = (co < Cout && tp < 9) ? w[((size_t)co * Cin + ci) * 9 + tp] : 0.0f;
        out[j] = (_Float16)v;
    }
    ((v16h*)wfrag)[(size_t)gid] = out;
}

// ---------------------------------------------------------------------------
// B-fragment builder: B[k, pixel] = xh[b, ci(k), y+dy(tap), x+dx(tap)].
// For Cin>=16 a lane's 16 consecutive K values share one tap -> one bounds
// check, one base pointer, 16 loads at immediate offsets j*HW.
// ---------------------------------------------------------------------------
__device__ __forceinline__ v16h load_bfrag(
    const _Float16* __restrict__ xh, int kb, int cinShift, int cinMask,
    bool pv, int b, int yy, int xx, int H, int W, int Cin)
{
    v16h bf;
    const int HW = H * W;
    if (cinShift >= 4) {
        int tp = kb >> cinShift;              // uniform over the 16-run
        bool ok = pv && (tp < 9);
        int sy = yy + (tp < 9 ? tp / 3 : 0) - 1;
        int sx = xx + (tp < 9 ? tp % 3 : 0) - 1;
        ok = ok && (unsigned)sy < (unsigned)H && (unsigned)sx < (unsigned)W;
        const _Float16* base =
            xh + (((size_t)b * Cin + (kb & cinMask)) * H + (ok ? sy : 0)) * W + (ok ? sx : 0);
#pragma unroll
        for (int j = 0; j < 16; ++j)
            bf[j] = ok ? base[(size_t)j * HW] : (_Float16)0.0f;
    } else {
#pragma unroll
        for (int j = 0; j < 16; ++j) {
            int k  = kb + j;
            int tp = k >> cinShift;
            _Float16 v = (_Float16)0.0f;
            if (pv && tp < 9) {
                int sy = yy + tp / 3 - 1;
                int sx = xx + tp % 3 - 1;
                if ((unsigned)sy < (unsigned)H && (unsigned)sx < (unsigned)W)
                    v = xh[(((size_t)b * Cin + (k & cinMask)) * H + sy) * W + sx];
            }
            bf[j] = v;
        }
    }
    return bf;
}

// ---------------------------------------------------------------------------
// 3x3 SAME conv as WMMA GEMM. 8 waves/block, one 16x16 (Cout x pixel) tile
// per wave, software-pipelined K loop with dual accumulators.
// B layout (16-bit 32x16): lane -> column n=l&15, K run (l>>4)*16.
// C/D layout: lane l -> n = l&15; vgpr r -> m = r + 8*(l>>4).
// ---------------------------------------------------------------------------
__global__ __launch_bounds__(256) void conv3x3_wmma(
    const _Float16* __restrict__ wfrag, const float* __restrict__ bias,
    const _Float16* __restrict__ xh, int Cin, int cinShift,
    float* __restrict__ out, int H, int W, int Cout, int Ktiles)
{
    const int lane  = threadIdx.x & 31;
    const int wid   = threadIdx.x >> 5;
    const int ntile = blockIdx.x * 8 + wid;
    const int mtile = blockIdx.y;
    const int HW = H * W;
    const int P  = 8 * HW;
    const int p  = ntile * 16 + (lane & 15);
    const bool pv = (p < P);
    int b = 0, yy = 0, xx = 0;
    if (pv) { b = p / HW; int r = p - b * HW; yy = r / W; xx = r - yy * W; }
    const int kb16 = (lane >> 4) * 16;
    const int cinMask = Cin - 1;
    const v16h* wf = ((const v16h*)wfrag) + (size_t)mtile * Ktiles * 32 + lane;

    v8f acc0 = {}, acc1 = {};
    // ---- software pipeline: preload tile 0 ----
    v16h a0 = wf[0];
    v16h b0 = load_bfrag(xh, kb16, cinShift, cinMask, pv, b, yy, xx, H, W, Cin);
    int kt = 0;
    for (; kt + 2 <= Ktiles; kt += 2) {
        // fetch tile kt+1 before consuming tile kt
        v16h a1 = wf[(size_t)(kt + 1) * 32];
        v16h b1 = load_bfrag(xh, (kt + 1) * 32 + kb16, cinShift, cinMask,
                             pv, b, yy, xx, H, W, Cin);
        acc0 = __builtin_amdgcn_wmma_f32_16x16x32_f16(
            false, a0, false, b0, (short)0, acc0, false, false);
        if (kt + 2 < Ktiles) {
            __builtin_prefetch(wf + (size_t)(kt + 3) * 32, 0, 3);
            a0 = wf[(size_t)(kt + 2) * 32];
            b0 = load_bfrag(xh, (kt + 2) * 32 + kb16, cinShift, cinMask,
                            pv, b, yy, xx, H, W, Cin);
        }
        acc1 = __builtin_amdgcn_wmma_f32_16x16x32_f16(
            false, a1, false, b1, (short)0, acc1, false, false);
    }
    if (kt < Ktiles) {   // odd Ktiles: last tile already resident in a0/b0
        acc0 = __builtin_amdgcn_wmma_f32_16x16x32_f16(
            false, a0, false, b0, (short)0, acc0, false, false);
    }
    v8f acc = acc0 + acc1;

    if (pv) {
#pragma unroll
        for (int r = 0; r < 8; ++r) {
            int co = mtile * 16 + r + 8 * (lane >> 4);
            if (co < Cout)
                out[(((size_t)b * Cout + co) * H + yy) * W + xx] = acc[r] + bias[co];
        }
    }
}

// ---------------------------------------------------------------------------
// f32 -> f16 conversion passes.
// ---------------------------------------------------------------------------
__global__ __launch_bounds__(256) void pack_f16(
    const float* __restrict__ x, _Float16* __restrict__ y, int total)
{
    int i = blockIdx.x * blockDim.x + threadIdx.x;
    if (i < total) y[i] = (_Float16)x[i];
}

// Copy skip [8,o,HW] (f32) into channels [o,2o) of cat buffer [8,2o,HW] (f16).
__global__ __launch_bounds__(256) void pack_skip_f16(
    const float* __restrict__ src, _Float16* __restrict__ dst,
    int o, int HW, int total)
{
    int i = blockIdx.x * blockDim.x + threadIdx.x;
    if (i >= total) return;
    int r = i % HW; int t = i / HW;
    int c = t % o;  int n = t / o;
    dst[((size_t)n * 2 * o + o + c) * HW + r] = (_Float16)src[i];
}

// ---------------------------------------------------------------------------
// BatchNorm (train mode, batch stats over N,H,W; biased var, eps=1e-5).
// ---------------------------------------------------------------------------
__global__ __launch_bounds__(256) void bn_stats(
    const float* __restrict__ x, float* __restrict__ mean, float* __restrict__ inv,
    int C, int HW)
{
    __shared__ float s1[256], s2[256];
    const int c = blockIdx.x;
    const int tid = threadIdx.x;
    const int total = 8 * HW;
    float a = 0.f, b = 0.f;
    for (int i = tid; i < total; i += 256) {
        int n = i / HW; int r = i - n * HW;
        float v = x[((size_t)n * C + c) * HW + r];
        a += v; b += v * v;
    }
    s1[tid] = a; s2[tid] = b;
    __syncthreads();
    for (int st = 128; st > 0; st >>= 1) {
        if (tid < st) { s1[tid] += s1[tid + st]; s2[tid] += s2[tid + st]; }
        __syncthreads();
    }
    if (tid == 0) {
        float m = s1[0] / (float)total;
        float v = s2[0] / (float)total - m * m;
        mean[c] = m;
        inv[c]  = rsqrtf(v + 1e-5f);
    }
}

__global__ __launch_bounds__(256) void bn_apply(
    const float* __restrict__ x, float* __restrict__ y,
    const float* __restrict__ mean, const float* __restrict__ inv,
    const float* __restrict__ g, const float* __restrict__ be,
    int C, int HW, int total, int relu)
{
    int i = blockIdx.x * blockDim.x + threadIdx.x;
    if (i >= total) return;
    int c = (i / HW) % C;
    float v = (x[i] - mean[c]) * inv[c] * g[c] + be[c];
    if (relu) v = fmaxf(v, 0.0f);
    y[i] = v;
}

__global__ __launch_bounds__(256) void maxpool2(
    const float* __restrict__ x, float* __restrict__ y,
    int C, int H, int W, int total)
{
    int i = blockIdx.x * blockDim.x + threadIdx.x;
    if (i >= total) return;
    int Wo = W >> 1, Ho = H >> 1;
    int xo = i % Wo; int t = i / Wo;
    int yo = t % Ho; t /= Ho;
    int c  = t % C;  int n = t / C;
    const float* p = x + ((size_t)(n * C + c) * H + yo * 2) * W + xo * 2;
    y[i] = fmaxf(fmaxf(p[0], p[1]), fmaxf(p[W], p[W + 1]));
}

// ---------------------------------------------------------------------------
// ConvTranspose2d(k=2,s=2) writing f16 directly into channels [0,o) of the
// cat buffer [8, 2o, 2H, 2W]:
//   cat[n,o_ch,2h+a,2w+b] = sum_c x[n,c,h,w]*W[c,o_ch,a,b] + B[o_ch]
// ---------------------------------------------------------------------------
__global__ __launch_bounds__(256) void convt2x2_h(
    const float* __restrict__ x, const float* __restrict__ w,
    const float* __restrict__ bias, _Float16* __restrict__ cat,
    int Cin, int Cout, int H, int W, int total)
{
    int i = blockIdx.x * blockDim.x + threadIdx.x;
    if (i >= total) return;
    int Wo = 2 * W, Ho = 2 * H;
    int X = i % Wo; int t = i / Wo;
    int Y = t % Ho; t /= Ho;
    int o = t % Cout; int n = t / Cout;
    int h = Y >> 1, a = Y & 1, wv = X >> 1, bq = X & 1;
    const float* xp = x + (size_t)n * Cin * H * W + (size_t)h * W + wv;
    float acc = bias[o];
    for (int c = 0; c < Cin; ++c)
        acc += xp[(size_t)c * H * W] * w[(((size_t)c * Cout + o) * 2 + a) * 2 + bq];
    cat[(((size_t)n * 2 * Cout + o) * Ho + Y) * Wo + X] = (_Float16)acc;
}

// ---------------------------------------------------------------------------
// 9-qubit statevector circuit. Gate schedule baked in (114 gates).
// kinds: 0=ry 1=rx 2=xx 3=xy 4=cry 5=crx.  wire w <-> bit (8 - w).
// ---------------------------------------------------------------------------
__constant__ unsigned char GK[114] = {
 0,0,0,0,0,0,0,0,0,  1,1,1,1,1,1,1,1,1,  2,2,2,2,2,2,2,2,  4,4,4,4,4,4,4,4,
 0,0,0,0,0,0,0,0,0,  3,3,3,3,3,3,3,3,    5,5,5,5,5,5,      1,1,1,1,1,1,1,1,1,
 2,2,2,2,2,2,2,2,    4,4,4,4,4,4,4,4,    0,0,0,0,0,0,0,0,0,3,3,3,3,3,3,3,3,
 5,5,5,5,5,5,        1,1,1,1,1,1,1,1,1 };
__constant__ unsigned char GA[114] = {
 0,1,2,3,4,5,6,7,8,  0,1,2,3,4,5,6,7,8,  0,8,1,7,2,6,3,5,  0,1,2,8,7,6,5,3,
 0,1,2,3,4,5,6,7,8,  4,4,3,5,2,6,1,7,    0,2,5,7,1,6,      0,1,2,3,4,5,6,7,8,
 0,8,1,7,2,6,3,5,    0,1,2,8,7,6,5,3,    0,1,2,3,4,5,6,7,8,4,4,3,5,2,6,1,7,
 0,2,5,7,1,6,        0,1,2,3,4,5,6,7,8 };
__constant__ unsigned char GB[114] = {
 0,0,0,0,0,0,0,0,0,  0,0,0,0,0,0,0,0,0,  1,7,2,6,3,5,4,4,  1,2,3,7,6,5,4,4,
 0,0,0,0,0,0,0,0,0,  3,5,2,6,1,7,0,8,    1,3,6,8,2,7,      0,0,0,0,0,0,0,0,0,
 1,7,2,6,3,5,4,4,    1,2,3,7,6,5,4,4,    0,0,0,0,0,0,0,0,0,3,5,2,6,1,7,0,8,
 1,3,6,8,2,7,        0,0,0,0,0,0,0,0,0 };

__global__ __launch_bounds__(256) void quantum_kernel(
    const float* __restrict__ p8, const float* __restrict__ qw,
    float* __restrict__ out)
{
    __shared__ float2 psi[512];
    __shared__ float  red[256];
    const int t   = threadIdx.x;
    const int blk = blockIdx.x;   // 0..63 = sample*8 + slice

    float v = tanhf(p8[blk * 256 + t]) * 1.5707963267948966f;
    red[t] = v * v;
    __syncthreads();
    for (int s = 128; s > 0; s >>= 1) {
        if (t < s) red[t] += red[t + s];
        __syncthreads();
    }
    float amp = v / (sqrtf(red[0]) + 1e-12f);
    // amplitude embedding on wires {0..3,5..8}; wire 4 starts in |0>:
    // insert a zero bit at position 4 of the 8-bit feature index.
    int e0 = ((t >> 4) << 5) | (t & 15);
    psi[e0]      = make_float2(amp, 0.0f);
    psi[e0 | 16] = make_float2(0.0f, 0.0f);
    __syncthreads();

    for (int g = 0; g < 114; ++g) {
        float th = qw[g] * 0.5f;
        float c, s;
        sincosf(th, &s, &c);
        int kind = GK[g];
        int b0   = 8 - GA[g];
        if (kind <= 1) {
            // single-qubit gate: 256 disjoint pairs, in-place
            int lm = (1 << b0) - 1;
            int i0 = ((t & ~lm) << 1) | (t & lm);
            int i1 = i0 | (1 << b0);
            float2 a0 = psi[i0], a1 = psi[i1], n0, n1;
            if (kind == 0) {              // ry: [[c,-s],[s,c]]
                n0 = make_float2(c * a0.x - s * a1.x, c * a0.y - s * a1.y);
                n1 = make_float2(s * a0.x + c * a1.x, s * a0.y + c * a1.y);
            } else {                       // rx: [[c,-is],[-is,c]]
                n0 = make_float2(c * a0.x + s * a1.y, c * a0.y - s * a1.x);
                n1 = make_float2(c * a1.x + s * a0.y, c * a1.y - s * a0.x);
            }
            psi[i0] = n0; psi[i1] = n1;
        } else if (t < 128) {
            // two-qubit gate: 128 disjoint quads, in-place
            int b1 = 8 - GB[g];
            int lo = (b0 < b1) ? b0 : b1;
            int hi = (b0 < b1) ? b1 : b0;
            int lml = (1 << lo) - 1, lmh = (1 << hi) - 1;
            int vI = ((t  & ~lml) << 1) | (t  & lml);
            vI     = ((vI & ~lmh) << 1) | (vI & lmh);
            int i00 = vI;
            int i01 = vI | (1 << b1);
            int i10 = vI | (1 << b0);
            int i11 = i10 | (1 << b1);
            float2 A0 = psi[i00], A1 = psi[i01], A2 = psi[i10], A3 = psi[i11];
            if (kind == 2) {              // xx: c*I - i s*antidiag
                psi[i00] = make_float2(c * A0.x + s * A3.y, c * A0.y - s * A3.x);
                psi[i01] = make_float2(c * A1.x + s * A2.y, c * A1.y - s * A2.x);
                psi[i10] = make_float2(c * A2.x + s * A1.y, c * A2.y - s * A1.x);
                psi[i11] = make_float2(c * A3.x + s * A0.y, c * A3.y - s * A0.x);
            } else if (kind == 3) {       // xy: middle block [[c, is],[is, c]]
                psi[i01] = make_float2(c * A1.x - s * A2.y, c * A1.y + s * A2.x);
                psi[i10] = make_float2(c * A2.x - s * A1.y, c * A2.y + s * A1.x);
            } else if (kind == 4) {       // cry: {2,3} block [[c,-s],[s,c]]
                psi[i10] = make_float2(c * A2.x - s * A3.x, c * A2.y - s * A3.y);
                psi[i11] = make_float2(s * A2.x + c * A3.x, s * A2.y + c * A3.y);
            } else {                      // crx: {2,3} block [[c,-is],[-is,c]]
                psi[i10] = make_float2(c * A2.x + s * A3.y, c * A2.y - s * A3.x);
                psi[i11] = make_float2(c * A3.x + s * A2.y, c * A3.y - s * A2.x);
            }
        }
        __syncthreads();
    }
    float2 a = psi[t];
    out[blk * 512 + t]       = a.x * a.x + a.y * a.y;
    a = psi[t + 256];
    out[blk * 512 + t + 256] = a.x * a.x + a.y * a.y;
}

// ---------------------------------------------------------------------------
// Final 1x1 projection: out[n,0,h,w] = sum_{c<4} y[n,c,h,w]*ow[c] + ob
// ---------------------------------------------------------------------------
__global__ __launch_bounds__(256) void final_proj(
    const float* __restrict__ y, const float* __restrict__ ow,
    const float* __restrict__ ob, float* __restrict__ out,
    int HW, int total)
{
    int i = blockIdx.x * blockDim.x + threadIdx.x;
    if (i >= total) return;
    int n = i / HW; int r = i - n * HW;
    const float* yp = y + (size_t)n * 4 * HW + r;
    out[i] = yp[0] * ow[0] + yp[(size_t)HW] * ow[1]
           + yp[(size_t)2 * HW] * ow[2] + yp[(size_t)3 * HW] * ow[3] + ob[0];
}

// ---------------------------------------------------------------------------
extern "C" void kernel_launch(void* const* d_in, const int* in_sizes, int n_in,
                              void* d_out, int out_size, void* d_ws, size_t ws_size,
                              hipStream_t stream)
{
    (void)in_sizes; (void)n_in; (void)out_size; (void)ws_size;

    // ---- workspace arena (~177 MB) ----
    char* ws = (char*)d_ws;
    size_t off = 0;
    auto afl = [&](size_t n) { float* p = (float*)(ws + off);
                               off = WS_ALIGN(off + n * sizeof(float)); return p; };

    static const int ECI[8] = {1, 4, 8, 16, 32, 64, 128, 256};
    static const int ECO[8] = {4, 8, 16, 32, 64, 128, 256, 512};
    static const int DCI[8] = {1024, 512, 256, 128, 64, 32, 16, 8};
    static const int DCO[8] = {512, 256, 128, 64, 32, 16, 8, 4};

    float* skipb[8];
    { int hh = 512;
      for (int L = 0; L < 8; ++L) { skipb[L] = afl((size_t)8 * ECO[L] * hh * hh); hh >>= 1; } }
    float* bufX  = afl(8388608);
    float* bufT  = afl(8388608);
    float* meanb = afl(1024);
    float* invb  = afl(1024);
    _Float16* bufH = (_Float16*)(ws + off);                     // f16 activations / cat
    off = WS_ALIGN(off + (size_t)16777216 * sizeof(_Float16));
    _Float16* wfrag = (_Float16*)(ws + off);                    // packed weights
    off = WS_ALIGN(off + (size_t)4718592 * sizeof(_Float16));   // max: dec0 (K=9216,M=512)

    // ================= Encoder =================
    const float* cur = (const float*)d_in[0];
    int h = 512;
    for (int L = 0; L < 8; ++L) {
        const int cin = ECI[L], cout = ECO[L];
        const int csh = __builtin_ctz(cin);
        const float* w   = (const float*)d_in[1 + 6 * L + 0];
        const float* bb  = (const float*)d_in[1 + 6 * L + 1];
        const float* g2  = (const float*)d_in[1 + 6 * L + 2];
        const float* be2 = (const float*)d_in[1 + 6 * L + 3];
        const float* g1  = (const float*)d_in[1 + 6 * L + 4];
        const float* be1 = (const float*)d_in[1 + 6 * L + 5];

        const int K = cin * 9, Kt = (K + 31) / 32, Mt = (cout + 15) / 16;
        const int wtot = Mt * Kt * 32;
        prep_wfrag<<<(wtot + 255) / 256, 256, 0, stream>>>(w, wfrag, cout, cin, csh, Mt, Kt);

        const int HW = h * h;
        const int itot = 8 * cin * HW;
        pack_f16<<<(itot + 255) / 256, 256, 0, stream>>>(cur, bufH, itot);

        const int Ntiles = 8 * HW / 16;
        conv3x3_wmma<<<dim3(Ntiles / 8, Mt), 256, 0, stream>>>(
            wfrag, bb, bufH, cin, csh, bufT, h, h, cout, Kt);

        const int tel = 8 * cout * HW;
        bn_stats<<<cout, 256, 0, stream>>>(bufT, meanb, invb, cout, HW);
        bn_apply<<<(tel + 255) / 256, 256, 0, stream>>>(
            bufT, bufT, meanb, invb, g2, be2, cout, HW, tel, 1);
        bn_stats<<<cout, 256, 0, stream>>>(bufT, meanb, invb, cout, HW);
        bn_apply<<<(tel + 255) / 256, 256, 0, stream>>>(
            bufT, skipb[L], meanb, invb, g1, be1, cout, HW, tel, 0);

        const int pt = tel / 4;
        maxpool2<<<(pt + 255) / 256, 256, 0, stream>>>(skipb[L], bufX, cout, h, h, pt);
        cur = bufX;
        h >>= 1;
    }

    // ================= Quantum bottleneck =================
    // bufX = p8 [8,512,2,2]; output probs [8,1024,2,2] -> bufT
    quantum_kernel<<<64, 256, 0, stream>>>(bufX, (const float*)d_in[113], bufT);

    // ================= Decoder =================
    float* yb = bufT;
    float* tb = bufX;
    h = 2;
    for (int i = 0; i < 8; ++i) {
        const int cin = DCI[i], o = DCO[i];
        const int cat = 2 * o;
        const int csh = __builtin_ctz(cat);
        const float* w   = (const float*)d_in[49 + 8 * i + 0];
        const float* bb  = (const float*)d_in[49 + 8 * i + 1];
        const float* g2  = (const float*)d_in[49 + 8 * i + 2];
        const float* be2 = (const float*)d_in[49 + 8 * i + 3];
        const float* g1  = (const float*)d_in[49 + 8 * i + 4];
        const float* be1 = (const float*)d_in[49 + 8 * i + 5];
        const float* uw  = (const float*)d_in[49 + 8 * i + 6];
        const float* ub  = (const float*)d_in[49 + 8 * i + 7];

        const int ho = 2 * h, HWo = ho * ho;
        // ConvTranspose -> f16 channels [0,o) of cat buffer
        const int ut = 8 * o * HWo;
        convt2x2_h<<<(ut + 255) / 256, 256, 0, stream>>>(yb, uw, ub, bufH, cin, o, h, h, ut);
        // skip -> f16 channels [o,2o) of cat buffer
        pack_skip_f16<<<(ut + 255) / 256, 256, 0, stream>>>(skipb[7 - i], bufH, o, HWo, ut);

        const int K = cat * 9, Kt = (K + 31) / 32, Mt = (o + 15) / 16;
        const int wtot = Mt * Kt * 32;
        prep_wfrag<<<(wtot + 255) / 256, 256, 0, stream>>>(w, wfrag, o, cat, csh, Mt, Kt);

        const int Ntiles = 8 * HWo / 16;
        conv3x3_wmma<<<dim3(Ntiles / 8, Mt), 256, 0, stream>>>(
            wfrag, bb, bufH, cat, csh, tb, ho, ho, o, Kt);

        const int tel = 8 * o * HWo;
        bn_stats<<<o, 256, 0, stream>>>(tb, meanb, invb, o, HWo);
        bn_apply<<<(tel + 255) / 256, 256, 0, stream>>>(
            tb, tb, meanb, invb, g2, be2, o, HWo, tel, 1);
        bn_stats<<<o, 256, 0, stream>>>(tb, meanb, invb, o, HWo);
        bn_apply<<<(tel + 255) / 256, 256, 0, stream>>>(
            tb, tb, meanb, invb, g1, be1, o, HWo, tel, 0);

        float* tmp = yb; yb = tb; tb = tmp;
        h = ho;
    }

    // ================= Output projection =================
    const int ft = 8 * 512 * 512;
    final_proj<<<(ft + 255) / 256, 256, 0, stream>>>(
        yb, (const float*)d_in[114], (const float*)d_in[115], (float*)d_out,
        512 * 512, ft);
}
